// Decoder_77163382440311
// MI455X (gfx1250) — matrix operand order repaired and checked by reading, not compiled, not run on hardware
//
#include <hip/hip_runtime.h>

// ---------------------------------------------------------------------------
// Types / helpers
// ---------------------------------------------------------------------------
typedef unsigned short u16;
typedef unsigned int   u32;
typedef __attribute__((ext_vector_type(16))) __bf16 v16bf;
typedef __attribute__((ext_vector_type(8)))  float  v8f;

union ABfrag { v16bf v; uint4 q[2]; };

__device__ __forceinline__ u16 f2bf(float f) {
    u32 u = __float_as_uint(f);
    u = (u + 0x7FFFu + ((u >> 16) & 1u)) >> 16;   // round-to-nearest-even
    return (u16)u;
}
__device__ __forceinline__ float sigf(float x) { return 1.0f / (1.0f + expf(-x)); }

// CDNA5 async global->LDS DMA (ISA 15.18.3 opcode 98): per-lane 16B copy,
// tracked by ASYNCcnt. vdst = LDS byte address (addr[31:0] of the generic
// shared pointer), vaddr = 64-bit global address.
__device__ __forceinline__ void async_lds_b128(u32 ldsoff, const void* g) {
    asm volatile("global_load_async_to_lds_b128 %0, %1, off"
                 :: "v"(ldsoff), "v"(g) : "memory");
}
__device__ __forceinline__ void wait_async0() {
    asm volatile("s_wait_asynccnt 0x0" ::: "memory");
}

// A-fragment (16x32 bf16, M x K). ISA 7.12.2: lanes 0-15 hold K {0..7,16..23},
// lanes 16-31 hold K {8..15,24..31}; row m = lane&15 within the tile.
__device__ __forceinline__ ABfrag loadA16(const u16* base, int row, int ldk, int k0, int lane) {
    ABfrag a; int klo = (lane & 16) ? 8 : 0;
    const u16* p = base + (size_t)row * ldk + k0 + klo;
    a.q[0] = *(const uint4*)p;
    a.q[1] = *(const uint4*)(p + 16);
    return a;
}
// B-fragment (32x16 bf16, K x N), column-major by lane: lanes 0-15 K {0..7,16..23},
// lanes 16-31 K {8..15,24..31}, col n = lane&15. Stored as [col][K] (row of weights).
__device__ __forceinline__ ABfrag loadB16(const u16* base, int col, int ldk, int k0, int lane) {
    ABfrag b; int kb = (lane & 16) ? 8 : 0;
    const u16* p = base + (size_t)col * ldk + k0 + kb;
    b.q[0] = *(const uint4*)p;
    b.q[1] = *(const uint4*)(p + 16);
    return b;
}
__device__ __forceinline__ v8f wmma_bf(ABfrag a, ABfrag b, v8f c) {
    return __builtin_amdgcn_wmma_f32_16x16x32_bf16(false, a.v, false, b.v, (short)0, c, false, false);
}

// ---------------------------------------------------------------------------
// Generic batched WMMA GEMM / conv1d-as-GEMM, templated on conv width KW so
// the k -> (channel d, tap tau) split folds to constant mul-shifts.
//   C[m, col] = sum_k A[m, k] * B[k, col]       k = d*KW + tau
//   A element addr = m*sAm + d*sAd + tau*sAt            (0 if d>=dlim or m>=M)
//   B element addr = pos*sBpos + d*sBd, pos=col+tau-pad (0 if OOB or d>=dlim)
// Interior plain-GEMM tiles stage via global_load_async_to_lds_b128
// (ASYNCcnt); conv / boundary tiles fall back to guarded element copies.
// Epilogue: t = acc + bias[idx];  idx = biasRow ? m : col
//   epi 1: t = sin(t)*e1[idx]      (prenet PSine)
//   epi 2: t = sin(t*e1+e2)*e3     (postnet BN + PSine)
//   epi 3: t = t*e1 + e2           (postnet output BN)
// Stores f32 to C and/or bf16 to Cbf at m*sCm + col*sCc.
// ---------------------------------------------------------------------------
template <int KW>
__global__ __launch_bounds__(256) void wmma_gemm_t(
    const u16* __restrict__ A, long sAm, long sAd, long sAt, long Abat, int dlim,
    const u16* __restrict__ B, long sBpos, long sBd, long Bbat, int poslim,
    float* C, u16* Cbf, long sCm, long sCc, long Cbat,
    int M, int Nc, int Ktiles, int pad,
    const float* bias, int biasRow, int epi,
    const float* e1, const float* e2, const float* e3)
{
    __shared__ __align__(16) u16 sA[64 * 32];   // [row][k]
    __shared__ __align__(16) u16 sB[64 * 32];   // [col][k]

    A += (long)blockIdx.z * Abat;
    B += (long)blockIdx.z * Bbat;
    if (C)   C   += (long)blockIdx.z * Cbat;
    if (Cbf) Cbf += (long)blockIdx.z * Cbat;

    const int mb = blockIdx.y * 64;
    const int nb = blockIdx.x * 64;
    const int tid = threadIdx.x;
    const int w = tid >> 5, lane = tid & 31;
    const int wm = w & 3, wn = w >> 2;          // 4 m-tiles x 2 col-halves
    const int kk = tid & 31;                    // staging column (constant/thread)
    const int r0 = tid >> 5;                    // staging row base
    const u32 sAoff = (u32)(uintptr_t)&sA[0];   // LDS byte offsets for async DMA
    const u32 sBoff = (u32)(uintptr_t)&sB[0];

    v8f acc[2];
    for (int nt = 0; nt < 2; ++nt)
        acc[nt] = (v8f){0.f,0.f,0.f,0.f,0.f,0.f,0.f,0.f};

    for (int kt = 0; kt < Ktiles; ++kt) {
        const int k0 = kt * 32;
        const int k = k0 + kk;
        const int d = k / KW;                   // constant divisor -> mul/shift
        const int tt = k - d * KW;
        const bool kok = (d < dlim);
        const long aoff = (long)d * sAd + (long)tt * sAt;
        const long boff = (long)d * sBd;
        const int posBase = nb + tt - pad;

        // Uniform fast-path predicates (plain GEMM, k-contiguous, in-bounds).
        const bool fastA = (KW == 1) && (sAd == 1) && (sAt == 0) &&
                           (mb + 64 <= M) && (k0 + 32 <= dlim);
        const bool fastB = (KW == 1) && (sBd == 1) && (pad == 0) &&
                           (nb + 64 <= poslim) && (k0 + 32 <= dlim);

        if (fastA) {            // 64 rows x 64B tile: 1 async b128 per thread
            int r = tid >> 2, ch = tid & 3;
            async_lds_b128(sAoff + (u32)(r * 64 + ch * 16),
                           A + (long)(mb + r) * sAm + k0 + ch * 8);
        } else {
            for (int j = 0; j < 8; ++j) {
                int r = r0 + 8 * j;
                int m = mb + r;
                u16 val = 0;
                if (kok && m < M) val = A[(long)m * sAm + aoff];
                sA[r * 32 + kk] = val;
            }
        }
        if (fastB) {
            int r = tid >> 2, ch = tid & 3;
            async_lds_b128(sBoff + (u32)(r * 64 + ch * 16),
                           B + (long)(nb + r) * sBpos + k0 + ch * 8);
        } else {
            for (int j = 0; j < 8; ++j) {
                int cB = r0 + 8 * j;
                int pos = posBase + cB;
                u16 val = 0;
                if (kok && pos >= 0 && pos < poslim) val = B[(long)pos * sBpos + boff];
                sB[cB * 32 + kk] = val;
            }
        }
        // Prefetch next k-tile's source lines (global_prefetch_b8) while the
        // WMMAs below run on the current tile.
        if (kt + 1 < Ktiles) {
            int k2 = k + 32;
            int d2 = k2 / KW;
            int tt2 = k2 - d2 * KW;
            if (d2 < dlim) {
                int m = mb + r0;
                if (m < M)
                    __builtin_prefetch(A + (long)m * sAm + (long)d2 * sAd + (long)tt2 * sAt, 0, 1);
                int pos2 = nb + tt2 - pad + r0;
                if (pos2 >= 0 && pos2 < poslim)
                    __builtin_prefetch(B + (long)pos2 * sBpos + (long)d2 * sBd, 0, 1);
            }
        }
        if (fastA || fastB) wait_async0();
        __syncthreads();
        ABfrag a = loadA16(sA, wm * 16 + (lane & 15), 32, 0, lane);
        for (int nt = 0; nt < 2; ++nt) {
            ABfrag b = loadB16(sB, wn * 32 + nt * 16 + (lane & 15), 32, 0, lane);
            acc[nt] = wmma_bf(a, b, acc[nt]);
        }
        __syncthreads();
    }

    for (int nt = 0; nt < 2; ++nt) {
        int col = nb + wn * 32 + nt * 16 + (lane & 15);
        for (int r = 0; r < 8; ++r) {
            int m = mb + wm * 16 + r + ((lane & 16) ? 8 : 0);
            if (m >= M || col >= Nc) continue;
            float t = acc[nt][r];
            int bi = biasRow ? m : col;
            if (bias) t += bias[bi];
            if (epi == 1)      t = sinf(t) * e1[bi];
            else if (epi == 2) t = sinf(t * e1[bi] + e2[bi]) * e3[bi];
            else if (epi == 3) t = t * e1[bi] + e2[bi];
            long off = (long)m * sCm + (long)col * sCc;
            if (C)   C[off] = t;
            if (Cbf) Cbf[off] = f2bf(t);
        }
    }
}

// ---------------------------------------------------------------------------
// Row softmax over 512 elements -> bf16 (feeds the a x q WMMA GEMM).
// grid.x = N*512 rows, 256 threads, 2 elements per thread.
// ---------------------------------------------------------------------------
__global__ __launch_bounds__(256) void softmax_row(const float* __restrict__ S,
                                                   u16* __restrict__ out)
{
    __shared__ float red[256];
    const float* p = S + (size_t)blockIdx.x * 512;
    u16* o = out + (size_t)blockIdx.x * 512;
    const int t = threadIdx.x;
    float a = p[t], b = p[t + 256];
    red[t] = fmaxf(a, b);
    __syncthreads();
    for (int s = 128; s > 0; s >>= 1) {
        if (t < s) red[t] = fmaxf(red[t], red[t + s]);
        __syncthreads();
    }
    float m = red[0];
    __syncthreads();
    float ea = expf(a - m), eb = expf(b - m);
    red[t] = ea + eb;
    __syncthreads();
    for (int s = 128; s > 0; s >>= 1) {
        if (t < s) red[t] += red[t + s];
        __syncthreads();
    }
    float inv = 1.0f / red[0];
    o[t] = f2bf(ea * inv);
    o[t + 256] = f2bf(eb * inv);
}

// ---------------------------------------------------------------------------
// Encoder LSTM step. grid.x = 2 (dir 0 fwd at s=t, dir 1 bwd at s=255-t).
// One 1024-thread block computes gates[32,1024] = Xp[s] + b + h @ Whh^T with
// WMMA into 128KB dynamic LDS (CDNA5 WGP has 320KB), then the pointwise update.
// ---------------------------------------------------------------------------
__global__ __launch_bounds__(1024) void enc_step(
    int t, int layer, int isLast,
    const float* __restrict__ XpF, const float* __restrict__ XpB,   // [S][32][1024]
    const u16* __restrict__ WhhAll,                                 // [4][1024][256] bf16
    const float* __restrict__ bAll,                                 // [4][1024]
    float* Hst, float* Cst,                                         // [4][32][256]
    u16* xOut,                                                      // [S][32][512] bf16
    u16* nct)                                                       // [32][512][256] bf16
{
    extern __shared__ char smem[];
    float* gates = (float*)smem;            // 32*1024 f32
    u16*   hA    = (u16*)(smem + 131072);   // 32*256 bf16

    const int dir = blockIdx.x;
    const int s = dir ? (255 - t) : t;
    const int sl = layer * 2 + dir;
    const u16*   W  = WhhAll + (size_t)sl * 1024 * 256;
    const float* bb = bAll + sl * 1024;
    float* h = Hst + sl * 8192;
    float* c = Cst + sl * 8192;
    const float* Xp = (dir ? XpB : XpF) + (size_t)s * 32768;

    const int tid = threadIdx.x;
    for (int i = tid; i < 8192; i += 1024) hA[i] = f2bf(h[i]);
    __syncthreads();

    const int w = tid >> 5, lane = tid & 31;
    const int wm = w & 1, ng = w >> 1;      // 2 m-tiles, 16 col groups of 4 tiles
    v8f acc[4];
    for (int nt = 0; nt < 4; ++nt) {
        int col = (ng * 4 + nt) * 16 + (lane & 15);
        for (int r = 0; r < 8; ++r) {
            int m = wm * 16 + r + ((lane & 16) ? 8 : 0);
            acc[nt][r] = Xp[m * 1024 + col] + bb[col];
        }
    }
    for (int kt = 0; kt < 8; ++kt) {
        int k0 = kt * 32;
        ABfrag a = loadA16(hA, wm * 16 + (lane & 15), 256, k0, lane);
        for (int nt = 0; nt < 4; ++nt) {
            int col = (ng * 4 + nt) * 16 + (lane & 15);
            ABfrag b = loadB16(W, col, 256, k0, lane);
            acc[nt] = wmma_bf(a, b, acc[nt]);
        }
    }
    for (int nt = 0; nt < 4; ++nt) {
        int col = (ng * 4 + nt) * 16 + (lane & 15);
        for (int r = 0; r < 8; ++r) {
            int m = wm * 16 + r + ((lane & 16) ? 8 : 0);
            gates[m * 1024 + col] = acc[nt][r];
        }
    }
    __syncthreads();
    for (int i = tid; i < 8192; i += 1024) {
        int n = i >> 8, j = i & 255;
        float ig = gates[n * 1024 + j];
        float fg = gates[n * 1024 + 256 + j];
        float gg = gates[n * 1024 + 512 + j];
        float og = gates[n * 1024 + 768 + j];
        float cc = sigf(fg) * c[i] + sigf(ig) * tanhf(gg);
        float hh = sigf(og) * tanhf(cc);
        c[i] = cc; h[i] = hh;
        u16 hb = f2bf(hh);
        xOut[((size_t)s * 32 + n) * 512 + dir * 256 + j] = hb;
        if (isLast) nct[((size_t)n * 512 + dir * 256 + j) * 256 + s] = hb;
    }
}

// ---------------------------------------------------------------------------
// Decoder LSTM step: 4 dependent layers chained inside one block.
// ---------------------------------------------------------------------------
__global__ __launch_bounds__(1024) void dec_step(
    int t,
    const float* __restrict__ Xp0,     // [32][256][1024] (layer0 input proj)
    const u16* __restrict__ WihR,      // [3][1024][256]
    const u16* __restrict__ Whh,       // [4][1024][256]
    const float* __restrict__ bAll,    // [4][1024]
    float* Hst, float* Cst,            // [4][32][256]
    u16* topBf,                        // [32][256][256]
    u16* stopIn)                       // [32][256][512] (writes cols 0:256)
{
    extern __shared__ char smem[];
    float* gates = (float*)smem;            // 128KB
    u16*   hA    = (u16*)(smem + 131072);   // 16KB
    u16*   hP    = (u16*)(smem + 147456);   // 16KB

    const int tid = threadIdx.x;
    const int w = tid >> 5, lane = tid & 31;
    const int wm = w & 1, ng = w >> 1;

    for (int l = 0; l < 4; ++l) {
        float* h = Hst + l * 8192;
        float* c = Cst + l * 8192;
        for (int i = tid; i < 8192; i += 1024) hA[i] = f2bf(h[i]);
        __syncthreads();

        const float* bb = bAll + l * 1024;
        v8f acc[4];
        for (int nt = 0; nt < 4; ++nt) {
            int col = (ng * 4 + nt) * 16 + (lane & 15);
            for (int r = 0; r < 8; ++r) {
                int m = wm * 16 + r + ((lane & 16) ? 8 : 0);
                float init = bb[col];
                if (l == 0) init += Xp0[((size_t)m * 256 + t) * 1024 + col];
                acc[nt][r] = init;
            }
        }
        if (l > 0) {   // input projection from previous layer's fresh h
            const u16* Wi = WihR + (size_t)(l - 1) * 1024 * 256;
            for (int kt = 0; kt < 8; ++kt) {
                int k0 = kt * 32;
                ABfrag a = loadA16(hP, wm * 16 + (lane & 15), 256, k0, lane);
                for (int nt = 0; nt < 4; ++nt) {
                    int col = (ng * 4 + nt) * 16 + (lane & 15);
                    ABfrag b = loadB16(Wi, col, 256, k0, lane);
                    acc[nt] = wmma_bf(a, b, acc[nt]);
                }
            }
        }
        const u16* Wh = Whh + (size_t)l * 1024 * 256;
        for (int kt = 0; kt < 8; ++kt) {
            int k0 = kt * 32;
            ABfrag a = loadA16(hA, wm * 16 + (lane & 15), 256, k0, lane);
            for (int nt = 0; nt < 4; ++nt) {
                int col = (ng * 4 + nt) * 16 + (lane & 15);
                ABfrag b = loadB16(Wh, col, 256, k0, lane);
                acc[nt] = wmma_bf(a, b, acc[nt]);
            }
        }
        for (int nt = 0; nt < 4; ++nt) {
            int col = (ng * 4 + nt) * 16 + (lane & 15);
            for (int r = 0; r < 8; ++r) {
                int m = wm * 16 + r + ((lane & 16) ? 8 : 0);
                gates[m * 1024 + col] = acc[nt][r];
            }
        }
        __syncthreads();
        for (int i = tid; i < 8192; i += 1024) {
            int n = i >> 8, j = i & 255;
            float ig = gates[n * 1024 + j];
            float fg = gates[n * 1024 + 256 + j];
            float gg = gates[n * 1024 + 512 + j];
            float og = gates[n * 1024 + 768 + j];
            float cc = sigf(fg) * c[i] + sigf(ig) * tanhf(gg);
            float hh = sigf(og) * tanhf(cc);
            c[i] = cc; h[i] = hh;
            u16 hb = f2bf(hh);
            hP[i] = hb;
            if (l == 3) {
                topBf[((size_t)n * 256 + t) * 256 + j] = hb;
                stopIn[((size_t)n * 256 + t) * 512 + j] = hb;
            }
        }
        __syncthreads();
    }
}

// ---------------------------------------------------------------------------
// Small utility kernels
// ---------------------------------------------------------------------------
__global__ void cvt_bf16(const float* __restrict__ s, u16* __restrict__ d, long n) {
    long i = (long)blockIdx.x * 256 + threadIdx.x;
    if (i < n) d[i] = f2bf(s[i]);
}
__global__ void zero_f32(float* p, long n) {
    long i = (long)blockIdx.x * 256 + threadIdx.x;
    if (i < n) p[i] = 0.0f;
}
// x0[s][n][e] = enc_out[n][s][e]  (bf16)
__global__ void transpose_x0(const float* __restrict__ enc, u16* __restrict__ x0) {
    long i = (long)blockIdx.x * 256 + threadIdx.x;
    if (i >= (long)256 * 32 * 512) return;
    int e = i & 511; long r = i >> 9;
    int n = r & 31; int s = r >> 5;
    x0[i] = f2bf(enc[((size_t)n * 256 + s) * 512 + e]);
}
// teacher[n][t][m] = (t==0) ? ic[m] : mels[n][m][t-1]  (bf16)
__global__ void build_teacher(const float* __restrict__ mels, const float* __restrict__ ic,
                              u16* __restrict__ tch) {
    long i = (long)blockIdx.x * 256 + threadIdx.x;
    if (i >= (long)32 * 256 * 80) return;
    int m = i % 80; long r = i / 80;
    int t = r & 255; int n = r >> 8;
    float v = (t == 0) ? ic[m] : mels[((size_t)n * 80 + m) * 256 + (t - 1)];
    tch[i] = f2bf(v);
}
// ctxNct[n][d][t] = ctx[n][t][d]  (bf16)
__global__ void ctx_transpose(const float* __restrict__ ctx, u16* __restrict__ o) {
    long i = (long)blockIdx.x * 256 + threadIdx.x;
    if (i >= (long)32 * 64 * 256) return;
    int t = i & 255; long r = i >> 8;
    int d = r & 63; int n = r >> 6;
    o[i] = f2bf(ctx[((size_t)n * 256 + t) * 64 + d]);
}
// decInBf = bf16(ctx + gamma * proj)
__global__ void dec_in_fuse(const float* __restrict__ ctx, const float* __restrict__ proj,
                            const float* __restrict__ gamma, u16* __restrict__ o, long n) {
    long i = (long)blockIdx.x * 256 + threadIdx.x;
    if (i < n) o[i] = f2bf(ctx[i] + gamma[0] * proj[i]);
}
// stopIn[n][t][256+j] = bf16(encH[2][n][j])   (enc layer1 fwd final hidden)
__global__ void stop_prefill(const float* __restrict__ Hst, u16* __restrict__ stopIn) {
    long i = (long)blockIdx.x * 256 + threadIdx.x;
    if (i >= (long)32 * 256 * 256) return;
    int j = i & 255; long r = i >> 8;
    int t = r & 255; int n = r >> 8;
    stopIn[((size_t)n * 256 + t) * 512 + 256 + j] = f2bf(Hst[2 * 8192 + n * 256 + j]);
}
__global__ void residual_add(const float* __restrict__ y5, const float* __restrict__ outs,
                             float* __restrict__ post, long n) {
    long i = (long)blockIdx.x * 256 + threadIdx.x;
    if (i < n) post[i] = y5[i] + outs[i];
}

// ---------------------------------------------------------------------------
// Host side
// ---------------------------------------------------------------------------
struct Bump { char* p; size_t off; };
template <typename T> static T* balloc(Bump& b, size_t n) {
    T* r = (T*)(b.p + b.off);
    b.off += ((n * sizeof(T)) + 255) & ~(size_t)255;
    return r;
}
static inline int cdiv(long a, long b) { return (int)((a + b - 1) / b); }

static void gemm(hipStream_t st,
                 const u16* A, long sAm, long sAd, long sAt, long Abat, int dlim,
                 const u16* B, long sBpos, long sBd, long Bbat, int poslim,
                 float* C, u16* Cbf, long sCm, long sCc, long Cbat,
                 int M, int Nc, int Ktiles, int KW, int pad, int batch,
                 const float* bias, int biasRow, int epi,
                 const float* e1, const float* e2, const float* e3) {
    dim3 g(cdiv(Nc, 64), cdiv(M, 64), batch);
    switch (KW) {
    case 5:
        wmma_gemm_t<5><<<g, 256, 0, st>>>(A, sAm, sAd, sAt, Abat, dlim,
                                          B, sBpos, sBd, Bbat, poslim,
                                          C, Cbf, sCm, sCc, Cbat,
                                          M, Nc, Ktiles, pad,
                                          bias, biasRow, epi, e1, e2, e3);
        break;
    case 9:
        wmma_gemm_t<9><<<g, 256, 0, st>>>(A, sAm, sAd, sAt, Abat, dlim,
                                          B, sBpos, sBd, Bbat, poslim,
                                          C, Cbf, sCm, sCc, Cbat,
                                          M, Nc, Ktiles, pad,
                                          bias, biasRow, epi, e1, e2, e3);
        break;
    case 13:
        wmma_gemm_t<13><<<g, 256, 0, st>>>(A, sAm, sAd, sAt, Abat, dlim,
                                           B, sBpos, sBd, Bbat, poslim,
                                           C, Cbf, sCm, sCc, Cbat,
                                           M, Nc, Ktiles, pad,
                                           bias, biasRow, epi, e1, e2, e3);
        break;
    default:
        wmma_gemm_t<1><<<g, 256, 0, st>>>(A, sAm, sAd, sAt, Abat, dlim,
                                          B, sBpos, sBd, Bbat, poslim,
                                          C, Cbf, sCm, sCc, Cbat,
                                          M, Nc, Ktiles, pad,
                                          bias, biasRow, epi, e1, e2, e3);
        break;
    }
}
static void cvt(hipStream_t st, const float* s, u16* d, long n) {
    cvt_bf16<<<cdiv(n, 256), 256, 0, st>>>(s, d, n);
}

extern "C" void kernel_launch(void* const* d_in, const int* in_sizes, int n_in,
                              void* d_out, int out_size, void* d_ws, size_t ws_size,
                              hipStream_t stream) {
    (void)in_sizes; (void)n_in; (void)out_size; (void)ws_size;
    const float* enc_out   = (const float*)d_in[0];
    const float* mels      = (const float*)d_in[1];
    const float* enc_Wih   = (const float*)d_in[5];
    const float* enc_Whh   = (const float*)d_in[6];
    const float* enc_b     = (const float*)d_in[7];
    const float* Wk        = (const float*)d_in[8];
    const float* bk        = (const float*)d_in[9];
    const float* Wv        = (const float*)d_in[10];
    const float* bv        = (const float*)d_in[11];
    const float* Wq        = (const float*)d_in[12];
    const float* bq        = (const float*)d_in[13];
    const float* ic        = (const float*)d_in[14];
    const float* pre_W1    = (const float*)d_in[15];
    const float* pre_b1    = (const float*)d_in[16];
    const float* pre_s1    = (const float*)d_in[17];
    const float* pre_W2    = (const float*)d_in[18];
    const float* pre_b2    = (const float*)d_in[19];
    const float* pre_s2    = (const float*)d_in[20];
    const float* pre_W3    = (const float*)d_in[21];
    const float* pre_b3    = (const float*)d_in[22];
    const float* proj_W    = (const float*)d_in[23];
    const float* proj_b    = (const float*)d_in[24];
    const float* gamma     = (const float*)d_in[25];
    const float* dec_Wih0  = (const float*)d_in[26];
    const float* dec_WihR  = (const float*)d_in[27];
    const float* dec_Whh   = (const float*)d_in[28];
    const float* dec_b     = (const float*)d_in[29];
    const float* fc_W      = (const float*)d_in[30];
    const float* fc_b      = (const float*)d_in[31];
    const float* stop_W    = (const float*)d_in[32];
    const float* stop_b    = (const float*)d_in[33];
    const float* pn_W0     = (const float*)d_in[34];
    const float* pn_b0     = (const float*)d_in[35];
    const float* pn_Wmid   = (const float*)d_in[36];
    const float* pn_bmid   = (const float*)d_in[37];
    const float* pn_W4     = (const float*)d_in[38];
    const float* pn_b4     = (const float*)d_in[39];
    const float* bn_s      = (const float*)d_in[40];
    const float* bn_b      = (const float*)d_in[41];
    const float* bn_so     = (const float*)d_in[42];
    const float* bn_bo     = (const float*)d_in[43];
    const float* psw       = (const float*)d_in[44];

    float* out_main = (float*)d_out;                // [32][80][256]
    float* out_post = out_main + 655360;            // [32][80][256]
    float* out_stop = out_main + 2 * 655360;        // [32][256]

    Bump bp{(char*)d_ws, 0};
    // bf16 weights
    u16* wEncIh = balloc<u16>(bp, 2097152);
    u16* wEncHh = balloc<u16>(bp, 1048576);
    u16* wK     = balloc<u16>(bp, 2359296);
    u16* wV     = balloc<u16>(bp, 2359296);
    u16* wQ     = balloc<u16>(bp, 786432);
    u16* wPre1  = balloc<u16>(bp, 10240);
    u16* wPre2  = balloc<u16>(bp, 16384);
    u16* wPre3  = balloc<u16>(bp, 8192);
    u16* wProj  = balloc<u16>(bp, 32768);
    u16* wDIh0  = balloc<u16>(bp, 65536);
    u16* wDIhR  = balloc<u16>(bp, 786432);
    u16* wDHh   = balloc<u16>(bp, 1048576);
    u16* wFc    = balloc<u16>(bp, 20480);
    u16* wStop  = balloc<u16>(bp, 512);
    u16* wPn0   = balloc<u16>(bp, 204800);
    u16* wPnM   = balloc<u16>(bp, 3932160);
    u16* wPn4   = balloc<u16>(bp, 204800);
    // activations
    u16*   x0     = balloc<u16>(bp, 4194304);   // [S][N][512]
    u16*   x1     = balloc<u16>(bp, 4194304);
    float* XpF    = balloc<float>(bp, 8388608); // [S][N][1024]; reused as scores
    float* XpB    = balloc<float>(bp, 8388608); // reused as decXp
    float* encH   = balloc<float>(bp, 32768);   // [4][32][256]
    float* encC   = balloc<float>(bp, 32768);
    u16*   nct    = balloc<u16>(bp, 4194304);   // [N][512][S]
    u16*   kBf    = balloc<u16>(bp, 4063232);   // [N][512][248]
    u16*   vBf    = balloc<u16>(bp, 4063232);
    u16*   aBf    = balloc<u16>(bp, 8388608);   // [N][512][512]
    u16*   tch    = balloc<u16>(bp, 655360);    // [N*T][80]
    u16*   p1     = balloc<u16>(bp, 1048576);
    u16*   p2     = balloc<u16>(bp, 1048576);
    float* ctx    = balloc<float>(bp, 524288);  // [N][T][64]
    u16*   ctxNct = balloc<u16>(bp, 524288);    // [N][64][T]
    u16*   qBf    = balloc<u16>(bp, 4194304);   // [N][512][T]
    u16*   oNct   = balloc<u16>(bp, 4194304);   // [N][512][T]
    float* decPrj = balloc<float>(bp, 524288);  // [N][T][64]
    u16*   decInB = balloc<u16>(bp, 524288);
    u16*   topBf  = balloc<u16>(bp, 2097152);   // [N][T][256]
    u16*   stopIn = balloc<u16>(bp, 4194304);   // [N][T][512]
    u16*   outBf  = balloc<u16>(bp, 655360);    // [N][80][T]
    u16*   pnA    = balloc<u16>(bp, 4194304);   // [N][512][T]
    u16*   pnB    = balloc<u16>(bp, 4194304);
    float* y5     = balloc<float>(bp, 655360);
    float* scores = XpF;                        // alias (encoder Xp dead by then)
    float* decXp  = XpB;                        // alias

    // ---- weight conversion ----
    cvt(stream, enc_Wih,  wEncIh, 2097152);
    cvt(stream, enc_Whh,  wEncHh, 1048576);
    cvt(stream, Wk,       wK,     2359296);
    cvt(stream, Wv,       wV,     2359296);
    cvt(stream, Wq,       wQ,     786432);
    cvt(stream, pre_W1,   wPre1,  10240);
    cvt(stream, pre_W2,   wPre2,  16384);
    cvt(stream, pre_W3,   wPre3,  8192);
    cvt(stream, proj_W,   wProj,  32768);
    cvt(stream, dec_Wih0, wDIh0,  65536);
    cvt(stream, dec_WihR, wDIhR,  786432);
    cvt(stream, dec_Whh,  wDHh,   1048576);
    cvt(stream, fc_W,     wFc,    20480);
    cvt(stream, stop_W,   wStop,  512);
    cvt(stream, pn_W0,    wPn0,   204800);
    cvt(stream, pn_Wmid,  wPnM,   3932160);
    cvt(stream, pn_W4,    wPn4,   204800);

    // ---- encoder ----
    transpose_x0<<<cdiv(4194304, 256), 256, 0, stream>>>(enc_out, x0);
    zero_f32<<<cdiv(32768, 256), 256, 0, stream>>>(encH, 32768);
    zero_f32<<<cdiv(32768, 256), 256, 0, stream>>>(encC, 32768);
    for (int l = 0; l < 2; ++l) {
        const u16* xin = l ? x1 : x0;
        u16* xout = l ? x0 : x1;   // layer1 xOut is a dummy sink
        gemm(stream, xin, 512, 1, 0, 0, 512,
             wEncIh + (size_t)(l * 2 + 0) * 524288, 512, 1, 0, 1024,
             XpF, nullptr, 1024, 1, 0, 8192, 1024, 16, 1, 0, 1,
             nullptr, 0, 0, nullptr, nullptr, nullptr);
        gemm(stream, xin, 512, 1, 0, 0, 512,
             wEncIh + (size_t)(l * 2 + 1) * 524288, 512, 1, 0, 1024,
             XpB, nullptr, 1024, 1, 0, 8192, 1024, 16, 1, 0, 1,
             nullptr, 0, 0, nullptr, nullptr, nullptr);
        for (int t = 0; t < 256; ++t)
            enc_step<<<2, 1024, 147456, stream>>>(t, l, (l == 1) ? 1 : 0,
                XpF, XpB, wEncHh, enc_b, encH, encC, xout, nct);
    }

    // ---- attention map ----
    gemm(stream, wK, 4608, 9, 1, 0, 512,  nct, 1, 256, 131072, 256,
         nullptr, kBf, 248, 1, 126976, 512, 248, 144, 9, 0, 32,
         bk, 1, 0, nullptr, nullptr, nullptr);
    gemm(stream, wV, 4608, 9, 1, 0, 512,  nct, 1, 256, 131072, 256,
         nullptr, vBf, 248, 1, 126976, 512, 248, 144, 9, 0, 32,
         bv, 1, 0, nullptr, nullptr, nullptr);
    gemm(stream, kBf, 248, 1, 0, 126976, 248,  vBf, 248, 1, 126976, 512,
         scores, nullptr, 512, 1, 262144, 512, 512, 8, 1, 0, 32,
         nullptr, 0, 0, nullptr, nullptr, nullptr);
    softmax_row<<<16384, 256, 0, stream>>>(scores, aBf);

    // ---- prenet + q ----
    build_teacher<<<cdiv(655360, 256), 256, 0, stream>>>(mels, ic, tch);
    gemm(stream, tch, 80, 1, 0, 0, 80,  wPre1, 80, 1, 0, 128,
         nullptr, p1, 128, 1, 0, 8192, 128, 3, 1, 0, 1,
         pre_b1, 0, 1, pre_s1, nullptr, nullptr);
    gemm(stream, p1, 128, 1, 0, 0, 128,  wPre2, 128, 1, 0, 128,
         nullptr, p2, 128, 1, 0, 8192, 128, 4, 1, 0, 1,
         pre_b2, 0, 1, pre_s2, nullptr, nullptr);
    gemm(stream, p2, 128, 1, 0, 0, 128,  wPre3, 128, 1, 0, 64,
         ctx, nullptr, 64, 1, 0, 8192, 64, 4, 1, 0, 1,
         pre_b3, 0, 0, nullptr, nullptr, nullptr);
    ctx_transpose<<<cdiv(524288, 256), 256, 0, stream>>>(ctx, ctxNct);
    gemm(stream, wQ, 1536, 24, 1, 0, 64,  ctxNct, 1, 256, 16384, 256,
         nullptr, qBf, 256, 1, 131072, 512, 256, 26, 13, 12, 32,
         bq, 1, 0, nullptr, nullptr, nullptr);
    gemm(stream, aBf, 512, 1, 0, 262144, 512,  qBf, 1, 256, 131072, 256,
         nullptr, oNct, 256, 1, 131072, 512, 256, 16, 1, 0, 32,
         nullptr, 0, 0, nullptr, nullptr, nullptr);
    gemm(stream, oNct, 1, 256, 0, 131072, 512,  wProj, 512, 1, 0, 64,
         decPrj, nullptr, 64, 1, 16384, 256, 64, 16, 1, 0, 32,
         proj_b, 0, 0, nullptr, nullptr, nullptr);
    dec_in_fuse<<<cdiv(524288, 256), 256, 0, stream>>>(ctx, decPrj, gamma, decInB, 524288);

    // ---- decoder ----
    gemm(stream, decInB, 64, 1, 0, 0, 64,  wDIh0, 64, 1, 0, 1024,
         decXp, nullptr, 1024, 1, 0, 8192, 1024, 2, 1, 0, 1,
         nullptr, 0, 0, nullptr, nullptr, nullptr);
    stop_prefill<<<cdiv(2097152, 256), 256, 0, stream>>>(encH, stopIn);
    for (int t = 0; t < 256; ++t)
        dec_step<<<1, 1024, 163840, stream>>>(t, decXp, wDIhR, wDHh, dec_b,
                                              encH, encC, topBf, stopIn);

    // ---- outputs / stop ----
    gemm(stream, topBf, 256, 1, 0, 65536, 256,  wFc, 256, 1, 0, 80,
         out_main, outBf, 1, 256, 20480, 256, 80, 8, 1, 0, 32,
         fc_b, 0, 0, nullptr, nullptr, nullptr);
    gemm(stream, stopIn, 512, 1, 0, 131072, 512,  wStop, 512, 1, 0, 1,
         out_stop, nullptr, 1, 1, 256, 256, 1, 16, 1, 0, 32,
         stop_b, 0, 0, nullptr, nullptr, nullptr);

    // ---- postnet ----
    gemm(stream, wPn0, 400, 5, 1, 0, 80,  outBf, 1, 256, 20480, 256,
         nullptr, pnA, 256, 1, 131072, 512, 256, 13, 5, 2, 32,
         pn_b0, 1, 2, bn_s, bn_b, psw);
    u16* pin = pnA; u16* pout = pnB;
    for (int i = 0; i < 3; ++i) {
        gemm(stream, wPnM + (size_t)i * 1310720, 2560, 5, 1, 0, 512,
             pin, 1, 256, 131072, 256,
             nullptr, pout, 256, 1, 131072, 512, 256, 80, 5, 2, 32,
             pn_bmid + i * 512, 1, 2, bn_s + 512 * (i + 1), bn_b + 512 * (i + 1), psw + 512 * (i + 1));
        u16* tp = pin; pin = pout; pout = tp;
    }
    gemm(stream, wPn4, 2560, 5, 1, 0, 512,  pin, 1, 256, 131072, 256,
         y5, nullptr, 256, 1, 20480, 80, 256, 80, 5, 2, 32,
         pn_b4, 1, 3, bn_so, bn_bo, nullptr);
    residual_add<<<cdiv(655360, 256), 256, 0, stream>>>(y5, out_main, out_post, 655360);
}
